// CompressiveDecoder_55611236549044
// MI455X (gfx1250) — matrix operand order repaired and checked by reading, not compiled, not run on hardware
//
#include <hip/hip_runtime.h>
#include <hip/hip_bf16.h>
#include <stdint.h>

// ---------------- problem constants ----------------
#define BATCH   2
#define SEQ     1024
#define DIM     1024
#define HEADS   16
#define DH      64
#define MEML    1024
#define CMEML   256
#define KVL     2304          // CMEML + MEML + SEQ
#define RATIO   4
#define SCALE   0.125f        // DH^-0.5

// d_out layout (floats): logits | new_mem | new_cmem | aux | weights
#define OFF_LOGITS   0
#define OFF_NEWMEM   (2*SEQ*DIM)                         // 2097152
#define OFF_NEWCMEM  (OFF_NEWMEM + 2*SEQ*DIM)            // 4194304
#define OFF_AUX      (OFF_NEWCMEM + 2*CMEML*DIM)         // 4718592
#define OFF_WEIGHTS  (OFF_AUX + 1)                       // 4718593 (4B-aligned only!)

// ---------------- WMMA types ----------------
typedef __bf16 bf16;
typedef __bf16 v16bf __attribute__((ext_vector_type(16)));
typedef __bf16 v8bf  __attribute__((ext_vector_type(8)));
typedef float  v8f   __attribute__((ext_vector_type(8)));

__device__ __forceinline__ v8f wmma_bf16(v16bf a, v16bf b, v8f c) {
  // v_wmma_f32_16x16x32_bf16  (8-arg form: neg_a, A, neg_b, B, c_mod, C, reuse_a, reuse_b)
  return __builtin_amdgcn_wmma_f32_16x16x32_bf16(false, a, false, b, (short)0, c, false, false);
}

__device__ __forceinline__ bf16 f2bf(float f) {
  unsigned u = __float_as_uint(f);
  unsigned r = (u + 0x7FFFu + ((u >> 16) & 1u)) >> 16;   // RNE truncate
  unsigned short s = (unsigned short)r;
  bf16 b; __builtin_memcpy(&b, &s, 2); return b;
}

// A fragment (16x32 bf16, M x K): lane&15 -> row, lane>=16 -> K+8 half; two 16B loads.
__device__ __forceinline__ v16bf afrag_load(const bf16* A, int lda, int m0, int k0) {
  int lane = threadIdx.x & 31;
  int row  = m0 + (lane & 15);
  int klo  = k0 + ((lane >> 4) << 3);
  const v8bf* p0 = (const v8bf*)(A + (size_t)row * lda + klo);
  const v8bf* p1 = (const v8bf*)(A + (size_t)row * lda + klo + 16);
  v8bf lo = *p0, hi = *p1;
  v16bf a;
#pragma unroll
  for (int j = 0; j < 8; ++j) { a[j] = lo[j]; a[j + 8] = hi[j]; }
  return a;
}

// B fragment (32x16 bf16, K x N), B row-major: lane&15 -> N, elems -> K (table order).
__device__ __forceinline__ v16bf bfrag_load(const bf16* B, int ldb, int k0, int n0) {
  int lane = threadIdx.x & 31;
  v16bf b;
#pragma unroll
  for (int e = 0; e < 16; ++e)
    b[e] = B[(size_t)(k0 + ((lane >> 4) << 3) + (e & 7) + ((e >> 3) << 4)) * ldb + n0 + (lane & 15)];
  return b;
}

__device__ __forceinline__ void zero8(v8f& v) {
#pragma unroll
  for (int i = 0; i < 8; ++i) v[i] = 0.f;
}

// CDNA5 async global->LDS copy of one 16-byte chunk per lane (ASYNCcnt +1 per wave).
__device__ __forceinline__ void async_copy_b128(unsigned lds_off, const void* gptr) {
  unsigned long long ga = (unsigned long long)gptr;
  asm volatile("global_load_async_to_lds_b128 %0, %1, off"
               :: "v"(lds_off), "v"(ga) : "memory");
}
#define WAIT_ASYNC_0() asm volatile("s_wait_asynccnt 0x0" ::: "memory")
#define WAIT_ASYNC_1() asm volatile("s_wait_asynccnt 0x1" ::: "memory")
#define WAIT_ASYNC_2() asm volatile("s_wait_asynccnt 0x2" ::: "memory")
#define SCHED_FENCE()  __builtin_amdgcn_sched_barrier(0)

// ---------------- pack / cast kernels ----------------
__global__ __launch_bounds__(256) void pack_kvin(const float* x, const float* mem,
                                                 const float* cmem, bf16* kvin,
                                                 float* newmem_out) {
  size_t i = (size_t)blockIdx.x * 256 + threadIdx.x;
  size_t total = (size_t)BATCH * KVL * DIM;
  if (i >= total) return;
  int e = (int)(i % DIM);
  size_t r = i / DIM;
  int j = (int)(r % KVL);
  int b = (int)(r / KVL);
  float v;
  if (j < CMEML)            v = cmem[((size_t)b * CMEML + j) * DIM + e];
  else if (j < CMEML + MEML) v = mem[((size_t)b * MEML + (j - CMEML)) * DIM + e];
  else {
    v = x[((size_t)b * SEQ + (j - CMEML - MEML)) * DIM + e];
    newmem_out[((size_t)b * SEQ + (j - CMEML - MEML)) * DIM + e] = v;  // new_mem = x
  }
  kvin[i] = f2bf(v);
}

__global__ __launch_bounds__(256) void cast_bf16(const float* s, bf16* d, int n) {
  int i = blockIdx.x * 256 + threadIdx.x;
  if (i < n) d[i] = f2bf(s[i]);
}

// conv_w [O=1024][I=1024][R=4] -> Wc[(r*1024+i)][o]  (K=4096 x N=1024 GEMM operand)
__global__ __launch_bounds__(256) void pack_convw(const float* cw, bf16* Wc) {
  int n = blockIdx.x * 256 + threadIdx.x;
  if (n >= 4096 * 1024) return;
  int o = n & 1023, row = n >> 10;
  int ii = row & 1023, r = row >> 10;
  Wc[n] = f2bf(cw[(size_t)o * 4096 + ii * 4 + r]);
}

__global__ void zero_scalar(float* p) { *p = 0.f; }

// ---------------- generic bf16 GEMM: C = alpha*A*B + bias ----------------
// block 256 = 8 waves: wave tile 32x64, block tile 128x128.
// B tiles (32x128) double-buffered in LDS via global_load_async_to_lds_b128;
// A fragments software-pipelined one K-step ahead; fragment loads fenced so the
// 8 WMMAs issue back-to-back behind a single wait.
__global__ __launch_bounds__(256) void gemm_bf16(const bf16* __restrict__ A, int lda,
                                                 const bf16* __restrict__ B, int ldb,
                                                 float* Cf, bf16* Cbf, int ldc,
                                                 const float* bias,
                                                 int M, int N, int K, float alpha) {
  __shared__ bf16 Bs[2][32][128];
  int tid = threadIdx.x;
  int w = tid >> 5, lane = tid & 31;
  int m0 = blockIdx.y * 128 + (w & 3) * 32;
  int n0b = blockIdx.x * 128;
  int nloc = (w >> 2) * 64;

  auto issue_b = [&](int kt, int buf) {
#pragma unroll
    for (int t = 0; t < 2; ++t) {
      int c = tid + t * 256;                 // 512 chunks of 8 bf16
      int row = c >> 4, cc = (c & 15) << 3;
      const bf16* g = B + (size_t)(kt * 32 + row) * ldb + n0b + cc;
      unsigned loff = (unsigned)(unsigned long long)&Bs[buf][row][cc];
      async_copy_b128(loff, g);
    }
  };

  v8f acc[2][4];
#pragma unroll
  for (int mi = 0; mi < 2; ++mi)
#pragma unroll
    for (int f = 0; f < 4; ++f) zero8(acc[mi][f]);

  int KT = K >> 5;                           // K / 32, K is a multiple of 32
  issue_b(0, 0);
  v16bf a_cur[2], a_nxt[2];
  a_cur[0] = afrag_load(A, lda, m0,      0);
  a_cur[1] = afrag_load(A, lda, m0 + 16, 0);
  a_nxt[0] = a_cur[0]; a_nxt[1] = a_cur[1];

  for (int kt = 0; kt < KT; ++kt) {
    int buf = kt & 1;
    bool more = (kt + 1 < KT);
    if (more) {
      issue_b(kt + 1, buf ^ 1);
      a_nxt[0] = afrag_load(A, lda, m0,      (kt + 1) << 5);
      a_nxt[1] = afrag_load(A, lda, m0 + 16, (kt + 1) << 5);
      WAIT_ASYNC_2();                        // tile kt done (async completes in order)
    } else {
      WAIT_ASYNC_0();
    }
    __syncthreads();
    v16bf bb[4];
#pragma unroll
    for (int f = 0; f < 4; ++f)
#pragma unroll
      for (int e = 0; e < 16; ++e)
        bb[f][e] = Bs[buf][((lane >> 4) << 3) + (e & 7) + ((e >> 3) << 4)][nloc + f * 16 + (lane & 15)];
    SCHED_FENCE();                           // all ds_loads before, all WMMAs after
#pragma unroll
    for (int f = 0; f < 4; ++f) acc[0][f] = wmma_bf16(a_cur[0], bb[f], acc[0][f]);
#pragma unroll
    for (int f = 0; f < 4; ++f) acc[1][f] = wmma_bf16(a_cur[1], bb[f], acc[1][f]);
    __syncthreads();                         // all waves done reading buf before reuse
    a_cur[0] = a_nxt[0]; a_cur[1] = a_nxt[1];
  }

  if (m0 >= M || n0b + nloc >= N) return;    // (grids are exact; kept for safety)
  int ncol = n0b + nloc + (lane & 15);
#pragma unroll
  for (int mi = 0; mi < 2; ++mi) {
    int mrow = m0 + mi * 16 + ((lane >> 4) << 3);
#pragma unroll
    for (int f = 0; f < 4; ++f)
#pragma unroll
      for (int r = 0; r < 8; ++r) {
        int m = mrow + r, n = ncol + f * 16;
        float v = acc[mi][f][r] * alpha + (bias ? bias[n] : 0.f);
        if (Cf)  Cf[(size_t)m * ldc + n] = v;
        if (Cbf) Cbf[(size_t)m * ldc + n] = f2bf(v);
      }
  }
}

// ---------------- attention: raw scores = scale * Q K^T ----------------
// 128 key rows x 64 dims staged in LDS via async copy; transposed gathers hit LDS.
__global__ __launch_bounds__(256) void attn_scores(const bf16* __restrict__ qb,
                                                   const bf16* __restrict__ kvb,
                                                   float* __restrict__ W) {
  __shared__ bf16 Ks[128][64];
  int bh = blockIdx.z, b = bh >> 4, h = bh & 15;
  const bf16* Q  = qb  + (size_t)b * SEQ * DIM + h * DH;            // lda DIM
  const bf16* Kp = kvb + (size_t)b * KVL * (2 * DIM) + h * DH;       // ldk 2*DIM
  int tid = threadIdx.x, w = tid >> 5, lane = tid & 31;
  int m0 = blockIdx.y * 64 + (w & 3) * 16;
  int n0b = blockIdx.x * 128;
  int nloc = (w >> 2) * 64;
#pragma unroll
  for (int t = 0; t < 4; ++t) {              // 1024 chunks: 8 per key row
    int c = tid + t * 256;
    int row = c >> 3, cc = (c & 7) << 3;
    const bf16* g = Kp + (size_t)(n0b + row) * (2 * DIM) + cc;
    unsigned loff = (unsigned)(unsigned long long)&Ks[row][cc];
    async_copy_b128(loff, g);
  }
  WAIT_ASYNC_0();
  __syncthreads();
  v8f acc[4];
#pragma unroll
  for (int f = 0; f < 4; ++f) zero8(acc[f]);
#pragma unroll
  for (int k0 = 0; k0 < DH; k0 += 32) {
    v16bf a = afrag_load(Q, DIM, m0, k0);
    v16bf bb[4];
#pragma unroll
    for (int f = 0; f < 4; ++f)
#pragma unroll
      for (int e = 0; e < 16; ++e)
        bb[f][e] = Ks[nloc + f * 16 + (lane & 15)][k0 + ((lane >> 4) << 3) + (e & 7) + ((e >> 3) << 4)];
    SCHED_FENCE();
#pragma unroll
    for (int f = 0; f < 4; ++f) acc[f] = wmma_bf16(a, bb[f], acc[f]);
  }
  float* Wb = W + (size_t)bh * SEQ * KVL;
  int mrow = m0 + ((lane >> 4) << 3), ncol = n0b + nloc + (lane & 15);
#pragma unroll
  for (int f = 0; f < 4; ++f)
#pragma unroll
    for (int r = 0; r < 8; ++r)
      Wb[(size_t)(mrow + r) * KVL + ncol + f * 16] = acc[f][r] * SCALE;
}

// ---------------- rel-pos: P = Q PE^T, scatter-added with shift ----------------
// shift(P)[m,n] = P[m, n + SEQ-1 - m]  =>  scores[m, j + m - (SEQ-1)] += scale*P[m,j]
__global__ __launch_bounds__(256) void attn_pos(const bf16* __restrict__ qb,
                                                const bf16* __restrict__ pe,
                                                float* __restrict__ W) {
  __shared__ bf16 Ps[128][64];
  int bh = blockIdx.z, b = bh >> 4, h = bh & 15;
  const bf16* Q  = qb + (size_t)b * SEQ * DIM + h * DH;
  const bf16* Pe = pe + (size_t)h * KVL * DH;                        // [2304][64]
  int tid = threadIdx.x, w = tid >> 5, lane = tid & 31;
  int m0 = blockIdx.y * 64 + (w & 3) * 16;
  int j0b = blockIdx.x * 128;
  int jloc = (w >> 2) * 64;
#pragma unroll
  for (int t = 0; t < 4; ++t) {
    int c = tid + t * 256;
    int row = c >> 3, cc = (c & 7) << 3;
    const bf16* g = Pe + (size_t)(j0b + row) * DH + cc;
    unsigned loff = (unsigned)(unsigned long long)&Ps[row][cc];
    async_copy_b128(loff, g);
  }
  WAIT_ASYNC_0();
  __syncthreads();
  v8f acc[4];
#pragma unroll
  for (int f = 0; f < 4; ++f) zero8(acc[f]);
#pragma unroll
  for (int k0 = 0; k0 < DH; k0 += 32) {
    v16bf a = afrag_load(Q, DIM, m0, k0);
    v16bf bb[4];
#pragma unroll
    for (int f = 0; f < 4; ++f)
#pragma unroll
      for (int e = 0; e < 16; ++e)
        bb[f][e] = Ps[jloc + f * 16 + (lane & 15)][k0 + ((lane >> 4) << 3) + (e & 7) + ((e >> 3) << 4)];
    SCHED_FENCE();
#pragma unroll
    for (int f = 0; f < 4; ++f) acc[f] = wmma_bf16(a, bb[f], acc[f]);
  }
  float* Wb = W + (size_t)bh * SEQ * KVL;
  int mrow = m0 + ((lane >> 4) << 3);
#pragma unroll
  for (int f = 0; f < 4; ++f)
#pragma unroll
    for (int r = 0; r < 8; ++r) {
      int m = mrow + r;
      int j = j0b + jloc + f * 16 + (lane & 15);
      int n = j + m - (SEQ - 1);                // unique (m,n) per (m,j): race-free RMW
      if ((unsigned)n < (unsigned)KVL) {
        size_t idx = (size_t)m * KVL + n;
        Wb[idx] += acc[f][r] * SCALE;
      }
    }
}

// ---------------- masked row softmax (in place on weights) ----------------
__global__ __launch_bounds__(256) void attn_softmax(float* __restrict__ W,
                                                    const unsigned char* __restrict__ im) {
  int row = blockIdx.x;                  // B*H*SEQ rows
  int bh = row / SEQ, m = row % SEQ, b = bh >> 4;
  float* Wr = W + (size_t)row * KVL;
  bool qv = im[b * SEQ + m] != 0;
  __shared__ float red[256];
  float vals[9];                         // 9*256 == 2304 exactly
  float lmax = -3.402823466e38f;
#pragma unroll
  for (int i = 0; i < 9; ++i) {
    int n = threadIdx.x + i * 256;
    float s = Wr[n];
    bool valid = (n < CMEML + MEML) ? true : (qv && im[b * SEQ + (n - CMEML - MEML)] != 0);
    if (!valid) s = -3.402823466e38f;
    vals[i] = s;
    lmax = fmaxf(lmax, s);
  }
  red[threadIdx.x] = lmax; __syncthreads();
  for (int s = 128; s > 0; s >>= 1) {
    if (threadIdx.x < s) red[threadIdx.x] = fmaxf(red[threadIdx.x], red[threadIdx.x + s]);
    __syncthreads();
  }
  float rmax = red[0]; __syncthreads();
  float lsum = 0.f;
#pragma unroll
  for (int i = 0; i < 9; ++i) { vals[i] = __expf(vals[i] - rmax); lsum += vals[i]; }
  red[threadIdx.x] = lsum; __syncthreads();
  for (int s = 128; s > 0; s >>= 1) {
    if (threadIdx.x < s) red[threadIdx.x] += red[threadIdx.x + s];
    __syncthreads();
  }
  float inv = 1.f / red[0];
#pragma unroll
  for (int i = 0; i < 9; ++i) Wr[threadIdx.x + i * 256] = vals[i] * inv;
}

// ---------------- out = attn @ V (merge-heads layout, bf16 out) ----------------
// V tiles (32x64) double-buffered in LDS via async copy.
__global__ __launch_bounds__(256) void attn_av(const float* __restrict__ W,
                                               const bf16* __restrict__ kvb,
                                               bf16* __restrict__ outb) {
  __shared__ bf16 Vs[2][32][64];
  int bh = blockIdx.y, b = bh >> 4, h = bh & 15;
  int tid = threadIdx.x, w = tid >> 5, lane = tid & 31;
  int m0 = blockIdx.x * 64 + (w & 3) * 16;
  int vc0 = (w >> 2) * 32;
  const float* Wr = W + (size_t)bh * SEQ * KVL;
  const bf16* Vp = kvb + (size_t)b * KVL * (2 * DIM) + DIM + h * DH;  // ld 2*DIM

  auto issue_v = [&](int kt, int buf) {      // 256 chunks: 8 per V row
    int row = tid >> 3, cc = (tid & 7) << 3;
    const bf16* g = Vp + (size_t)(kt * 32 + row) * (2 * DIM) + cc;
    unsigned loff = (unsigned)(unsigned long long)&Vs[buf][row][cc];
    async_copy_b128(loff, g);
  };

  v8f acc[2]; zero8(acc[0]); zero8(acc[1]);
  issue_v(0, 0);
  const int KT = KVL / 32;                    // 72
  for (int kt = 0; kt < KT; ++kt) {
    int buf = kt & 1;
    bool more = (kt + 1 < KT);
    if (more) issue_v(kt + 1, buf ^ 1);
    int row = m0 + (lane & 15);
    int klo = (kt << 5) + ((lane >> 4) << 3);
    const float* p = Wr + (size_t)row * KVL + klo;
    if (more) __builtin_prefetch(p + 32);
    v16bf a;
#pragma unroll
    for (int j = 0; j < 8; ++j) { a[j] = f2bf(p[j]); a[j + 8] = f2bf(p[16 + j]); }
    if (more) WAIT_ASYNC_1(); else WAIT_ASYNC_0();
    __syncthreads();
    v16bf bb[2];
#pragma unroll
    for (int f = 0; f < 2; ++f)
#pragma unroll
      for (int e = 0; e < 16; ++e)
        bb[f][e] = Vs[buf][((lane >> 4) << 3) + (e & 7) + ((e >> 3) << 4)][vc0 + f * 16 + (lane & 15)];
    SCHED_FENCE();
#pragma unroll
    for (int f = 0; f < 2; ++f) acc[f] = wmma_bf16(a, bb[f], acc[f]);
    __syncthreads();
  }
  int mrow = m0 + ((lane >> 4) << 3);
#pragma unroll
  for (int f = 0; f < 2; ++f)
#pragma unroll
    for (int r = 0; r < 8; ++r)
      outb[((size_t)b * SEQ + mrow + r) * DIM + h * DH + vc0 + f * 16 + (lane & 15)] =
          f2bf(acc[f][r]);
}

// ---------------- flash attention (aux loss), online softmax per 16-row wave tile ----
// K tiles (64x64) double-buffered in LDS via async copy; V frags stream from global.
__global__ __launch_bounds__(256) void flash_attn(const bf16* __restrict__ qb,
                                                  const bf16* __restrict__ Kbase,
                                                  const bf16* __restrict__ Vbase,
                                                  size_t strideB, int ldk, int kvlen,
                                                  float* __restrict__ O) {
  int bh = blockIdx.y, b = bh >> 4, h = bh & 15;
  int tid = threadIdx.x, w = tid >> 5, lane = tid & 31;
  int m0 = blockIdx.x * 128 + w * 16;
  const bf16* Q  = qb + (size_t)b * SEQ * DIM + h * DH;
  const bf16* Kp = Kbase + (size_t)b * strideB + h * DH;
  const bf16* Vp = Vbase + (size_t)b * strideB + h * DH;
  __shared__ bf16 pbuf[8][16][64];
  __shared__ bf16 Ks[2][64][64];

  auto issue_k = [&](int jt, int buf) {      // 512 chunks: 8 per key row, 2/thread
#pragma unroll
    for (int t = 0; t < 2; ++t) {
      int c = tid + t * 256;
      int row = c >> 3, cc = (c & 7) << 3;
      const bf16* g = Kp + (size_t)(jt * 64 + row) * ldk + cc;
      unsigned loff = (unsigned)(unsigned long long)&Ks[buf][row][cc];
      async_copy_b128(loff, g);
    }
  };

  v8f o[4];
#pragma unroll
  for (int f = 0; f < 4; ++f) zero8(o[f]);
  float rmax[8], rsum[8];
#pragma unroll
  for (int r = 0; r < 8; ++r) { rmax[r] = -3.402823466e38f; rsum[r] = 0.f; }

  int JT = kvlen >> 6;                       // kvlen / 64
  issue_k(0, 0);
  for (int jt = 0; jt < JT; ++jt) {
    int buf = jt & 1;
    bool more = (jt + 1 < JT);
    if (more) { issue_k(jt + 1, buf ^ 1); WAIT_ASYNC_2(); }
    else      { WAIT_ASYNC_0(); }
    __syncthreads();

    v8f s[4];
#pragma unroll
    for (int f = 0; f < 4; ++f) zero8(s[f]);
#pragma unroll
    for (int k0 = 0; k0 < DH; k0 += 32) {
      v16bf a = afrag_load(Q, DIM, m0, k0);
      v16bf bb[4];
#pragma unroll
      for (int f = 0; f < 4; ++f)
#pragma unroll
        for (int e = 0; e < 16; ++e)
          bb[f][e] = Ks[buf][f * 16 + (lane & 15)][k0 + ((lane >> 4) << 3) + (e & 7) + ((e >> 3) << 4)];
      SCHED_FENCE();
#pragma unroll
      for (int f = 0; f < 4; ++f) s[f] = wmma_bf16(a, bb[f], s[f]);
    }
#pragma unroll
    for (int f = 0; f < 4; ++f)
#pragma unroll
      for (int r = 0; r < 8; ++r) s[f][r] *= SCALE;

#pragma unroll
    for (int r = 0; r < 8; ++r) {
      float mx = fmaxf(fmaxf(s[0][r], s[1][r]), fmaxf(s[2][r], s[3][r]));
#pragma unroll
      for (int off = 1; off < 16; off <<= 1) mx = fmaxf(mx, __shfl_xor(mx, off, 32));
      float nm = fmaxf(rmax[r], mx);
      float fac = __expf(rmax[r] - nm);
      rsum[r] *= fac;
#pragma unroll
      for (int f = 0; f < 4; ++f) o[f][r] *= fac;
      rmax[r] = nm;
      float ps = 0.f;
#pragma unroll
      for (int f = 0; f < 4; ++f) { float p = __expf(s[f][r] - nm); s[f][r] = p; ps += p; }
#pragma unroll
      for (int off = 1; off < 16; off <<= 1) ps += __shfl_xor(ps, off, 32);
      rsum[r] += ps;
    }
    // probs C-layout -> LDS -> A-layout for P@V
    int hi8 = (lane >> 4) << 3;
#pragma unroll
    for (int f = 0; f < 4; ++f)
#pragma unroll
      for (int r = 0; r < 8; ++r)
        pbuf[w][r + hi8][f * 16 + (lane & 15)] = f2bf(s[f][r]);
    __syncthreads();
#pragma unroll
    for (int ks = 0; ks < 64; ks += 32) {
      int row = lane & 15;
      int kl = ks + hi8;
      v16bf a;
#pragma unroll
      for (int j = 0; j < 8; ++j) { a[j] = pbuf[w][row][kl + j]; a[j + 8] = pbuf[w][row][kl + 16 + j]; }
#pragma unroll
      for (int f = 0; f < 4; ++f) {
        v16bf bv = bfrag_load(Vp, ldk, jt * 64 + ks, f * 16);
        o[f] = wmma_bf16(a, bv, o[f]);
      }
    }
    __syncthreads();
  }
  int mrow = m0 + ((lane >> 4) << 3);
#pragma unroll
  for (int r = 0; r < 8; ++r) {
    float inv = 1.f / rsum[r];
#pragma unroll
    for (int f = 0; f < 4; ++f)
      O[((size_t)bh * SEQ + mrow + r) * DH + f * 16 + (lane & 15)] = o[f][r] * inv;
  }
}

// ---------------- aux loss: mean((O1-O2)^2) ----------------
__global__ __launch_bounds__(256) void aux_reduce(const float* O1, const float* O2,
                                                  float* aux, int n, float invn) {
  __shared__ float red[256];
  float s = 0.f;
#pragma unroll
  for (int t = 0; t < 4; ++t) {
    int idx = blockIdx.x * 1024 + t * 256 + threadIdx.x;
    if (idx < n) { float d = O1[idx] - O2[idx]; s += d * d; }
  }
  red[threadIdx.x] = s; __syncthreads();
  for (int k = 128; k > 0; k >>= 1) {
    if (threadIdx.x < k) red[threadIdx.x] += red[threadIdx.x + k];
    __syncthreads();
  }
  if (threadIdx.x == 0) atomicAdd(aux, red[0] * invn);
}

// ---------------- host launcher ----------------
extern "C" void kernel_launch(void* const* d_in, const int* in_sizes, int n_in,
                              void* d_out, int out_size, void* d_ws, size_t ws_size,
                              hipStream_t stream) {
  const float* x       = (const float*)d_in[0];
  const float* mem     = (const float*)d_in[1];
  const float* cmem    = (const float*)d_in[2];
  const float* pos_emb = (const float*)d_in[3];
  const unsigned char* imask = (const unsigned char*)d_in[4];  // bool array
  const float* Wq      = (const float*)d_in[5];
  const float* Wkv     = (const float*)d_in[6];
  const float* Wout    = (const float*)d_in[7];
  const float* bout    = (const float*)d_in[8];
  const float* conv_w  = (const float*)d_in[9];
  const float* conv_b  = (const float*)d_in[10];

  float* out = (float*)d_out;
  float* out_logits  = out + OFF_LOGITS;
  float* out_newmem  = out + OFF_NEWMEM;
  float* out_newcmem = out + OFF_NEWCMEM;
  float* out_aux     = out + OFF_AUX;
  float* out_weights = out + OFF_WEIGHTS;

  // workspace layout (bytes, all 256B aligned)
  char* ws = (char*)d_ws;
  bf16* kvinB = (bf16*)(ws + 0);                    //  9.4 MB [B][KVL][DIM]
  bf16* qB    = (bf16*)(ws + 9437184);              //  4.2 MB [B][SEQ][DIM]
  bf16* kvB   = (bf16*)(ws + 13631488);             // 18.9 MB [B][KVL][2*DIM]
  bf16* WqB   = (bf16*)(ws + 32505856);
  bf16* WkvB  = (bf16*)(ws + 34603008);
  bf16* WoutB = (bf16*)(ws + 38797312);
  bf16* WcB   = (bf16*)(ws + 40894464);             //  8.4 MB [4096][1024]
  bf16* peB   = (bf16*)(ws + 49283072);             //  4.7 MB [H][KVL][DH]
  bf16* cB    = (bf16*)(ws + 54001664);             //  1.0 MB [B][CMEML][DIM]
  bf16* ckvB  = (bf16*)(ws + 55050240);             //  2.1 MB [B][CMEML][2*DIM]
  bf16* outB  = (bf16*)(ws + 57147392);             //  4.2 MB [B][SEQ][DIM]
  float* O1   = (float*)(ws + 61341696);            // 16.8 MB [B*H][SEQ][DH]
  float* O2   = (float*)(ws + 78118912);            // 16.8 MB

  // 1) packing
  pack_kvin<<<(BATCH * KVL * DIM + 255) / 256, 256, 0, stream>>>(x, mem, cmem, kvinB, out_newmem);
  cast_bf16<<<(DIM * DIM + 255) / 256, 256, 0, stream>>>(Wq, WqB, DIM * DIM);
  cast_bf16<<<(DIM * 2 * DIM + 255) / 256, 256, 0, stream>>>(Wkv, WkvB, DIM * 2 * DIM);
  cast_bf16<<<(DIM * DIM + 255) / 256, 256, 0, stream>>>(Wout, WoutB, DIM * DIM);
  cast_bf16<<<(HEADS * KVL * DH + 255) / 256, 256, 0, stream>>>(pos_emb, peB, HEADS * KVL * DH);
  pack_convw<<<(4096 * 1024 + 255) / 256, 256, 0, stream>>>(conv_w, WcB);

  // 2) projections (per batch, all WMMA GEMMs; block tile 128x128)
  for (int b = 0; b < BATCH; ++b) {
    // q = x @ Wq  (bf16 out only)
    gemm_bf16<<<dim3(DIM / 128, SEQ / 128), 256, 0, stream>>>(
        kvinB + ((size_t)b * KVL + CMEML + MEML) * DIM, DIM, WqB, DIM,
        nullptr, qB + (size_t)b * SEQ * DIM, DIM, nullptr, SEQ, DIM, DIM, 1.f);
    // kv = [cmem;mem;x] @ Wkv
    gemm_bf16<<<dim3(2 * DIM / 128, KVL / 128), 256, 0, stream>>>(
        kvinB + (size_t)b * KVL * DIM, DIM, WkvB, 2 * DIM,
        nullptr, kvB + (size_t)b * KVL * 2 * DIM, 2 * DIM, nullptr, KVL, 2 * DIM, DIM, 1.f);
    // compressed = conv(mem) as GEMM  (K = 4096), -> new_cmem (f32) + bf16 copy
    gemm_bf16<<<dim3(DIM / 128, CMEML / 128), 256, 0, stream>>>(
        kvinB + ((size_t)b * KVL + CMEML) * DIM, RATIO * DIM, WcB, DIM,
        out_newcmem + (size_t)b * CMEML * DIM, cB + (size_t)b * CMEML * DIM, DIM,
        conv_b, CMEML, DIM, RATIO * DIM, 1.f);
  }
  for (int b = 0; b < BATCH; ++b) {
    // ckv = compressed @ Wkv
    gemm_bf16<<<dim3(2 * DIM / 128, CMEML / 128), 256, 0, stream>>>(
        cB + (size_t)b * CMEML * DIM, DIM, WkvB, 2 * DIM,
        nullptr, ckvB + (size_t)b * CMEML * 2 * DIM, 2 * DIM, nullptr,
        CMEML, 2 * DIM, DIM, 1.f);
  }

  // 3) attention
  attn_scores<<<dim3(KVL / 128, SEQ / 64, BATCH * HEADS), 256, 0, stream>>>(qB, kvB, out_weights);
  attn_pos<<<dim3(KVL / 128, SEQ / 64, BATCH * HEADS), 256, 0, stream>>>(qB, peB, out_weights);
  attn_softmax<<<BATCH * HEADS * SEQ, 256, 0, stream>>>(out_weights, imask);
  attn_av<<<dim3(SEQ / 64, BATCH * HEADS), 256, 0, stream>>>(out_weights, kvB, outB);

  // 4) logits = out @ Wout + bout
  for (int b = 0; b < BATCH; ++b)
    gemm_bf16<<<dim3(DIM / 128, SEQ / 128), 256, 0, stream>>>(
        outB + (size_t)b * SEQ * DIM, DIM, WoutB, DIM,
        out_logits + (size_t)b * SEQ * DIM, nullptr, DIM, bout, SEQ, DIM, DIM, 1.f);

  // 5) aux loss: two flash attentions + squared-diff mean
  zero_scalar<<<1, 1, 0, stream>>>(out_aux);
  // old-mem keys/values = kv rows [CMEML, CMEML+MEML)
  flash_attn<<<dim3(SEQ / 128, BATCH * HEADS), 256, 0, stream>>>(
      qB, kvB + (size_t)CMEML * 2 * DIM, kvB + (size_t)CMEML * 2 * DIM + DIM,
      (size_t)KVL * 2 * DIM, 2 * DIM, MEML, O1);
  // compressed keys/values
  flash_attn<<<dim3(SEQ / 128, BATCH * HEADS), 256, 0, stream>>>(
      qB, ckvB, ckvB + DIM, (size_t)CMEML * 2 * DIM, 2 * DIM, CMEML, O2);
  int naux = BATCH * HEADS * SEQ * DH;
  aux_reduce<<<(naux + 1023) / 1024, 256, 0, stream>>>(O1, O2, out_aux, naux, 1.f / naux);
}
